// ActorCriticRNN_57277683860106
// MI455X (gfx1250) — matrix operand order repaired and checked by reading, not compiled, not run on hardware
//
#include <hip/hip_runtime.h>

typedef __attribute__((ext_vector_type(16))) _Float16 v16h;
typedef __attribute__((ext_vector_type(8)))  float    v8f;

#define T_DIM 128
#define B_DIM 128
#define HID 128
#define PIX 54              // 6*9
#define NPIX (T_DIM*B_DIM*PIX)
#define TB (T_DIM*B_DIM)

__device__ __forceinline__ v8f v8f_zero() { v8f z = {}; return z; }

__device__ __forceinline__ v8f wmma_f16(v16h a, v16h b, v8f c) {
  return __builtin_amdgcn_wmma_f32_16x16x32_f16(false, a, false, b, (short)0, c, false, false);
}

// Build the A operand (16x32 f16, contiguous k = 16h + s) from a per-k generator.
// lanes 0-15: M=lane, K = 0..15; lanes 16-31: M=lane-16, K = 16..31
template <typename F>
__device__ __forceinline__ v16h gather_a(F f) {
  const int lane = threadIdx.x & 31;
  const int koff = ((lane >> 4) << 4); // 0 or 16
  v16h a;
#pragma unroll
  for (int v = 0; v < 16; ++v) {
    a[v] = f(koff + v);
  }
  return a;
}

// A operand from row-major f16 buffer (tile origin p, row stride ld halves)
__device__ __forceinline__ v16h load_a_rm(const _Float16* p, int ld) {
  const int lane = threadIdx.x & 31;
  const _Float16* row = p + (long)(lane & 15) * ld;
  return gather_a([&](int k) -> _Float16 { return row[k]; });
}

// B operand (32x16 f16) from K-major (transposed) weights WT[n][k]:
// tile origin p = WT + n0*ldk + k0.  lane = N, 16 contiguous K halves.
__device__ __forceinline__ v16h load_b_km(const _Float16* p, int ldk) {
  const int lane = threadIdx.x & 31;
  const _Float16* col = p + (long)(lane & 15) * ldk + ((lane >> 4) << 4);
  v16h b;
#pragma unroll
  for (int v = 0; v < 16; ++v) b[v] = col[v];
  return b;
}

// ---------------------------------------------------------------- prep: f16 transposed weights
__global__ void k_prep(const float* c1, const float* c2, const float* c3,
                       const float* c4, const float* c5, const float* c6,
                       const float* we, const float* wgi, const float* wgh,
                       const float* wa1, const float* wa2, const float* wv1, const float* wv2,
                       _Float16* W1T, _Float16* W2T, _Float16* W3T,
                       _Float16* W4T, _Float16* W5T, _Float16* W6T,
                       _Float16* WembT, _Float16* WginT, _Float16* WghhT,
                       _Float16* Wa1T, _Float16* Wa2T, _Float16* Wv1T, _Float16* Wv2T) {
  const int tid = blockIdx.x * blockDim.x + threadIdx.x;
  const int stride = gridDim.x * blockDim.x;
  for (int i = tid; i < 128*32; i += stride) {           // [128n][32k], K pad 26->32
    int n = i >> 5, k = i & 31;
    W1T[i] = (_Float16)(k < 26 ? c1[k*128 + n] : 0.f);
  }
  for (int i = tid; i < 128*128; i += stride) {
    int n = i >> 7, k = i & 127;
    W2T[i] = (_Float16)c2[k*128 + n];
  }
  for (int i = tid; i < 16*128; i += stride) {           // N pad 8->16
    int n = i >> 7, k = i & 127;
    W3T[i] = (_Float16)(n < 8 ? c3[k*8 + n] : 0.f);
  }
  for (int i = tid; i < 16*96; i += stride) {            // im2col K=72 pad 96
    int n = i / 96, k = i % 96;
    W4T[i] = (_Float16)(k < 72 ? c4[k*16 + n] : 0.f);
  }
  for (int i = tid; i < 32*160; i += stride) {           // K=144 pad 160
    int n = i / 160, k = i % 160;
    W5T[i] = (_Float16)(k < 144 ? c5[k*32 + n] : 0.f);
  }
  for (int i = tid; i < 32*288; i += stride) {           // K=288
    int n = i / 288, k = i % 288;
    W6T[i] = (_Float16)c6[k*32 + n];
  }
  for (long i = tid; i < 128L*1728; i += stride) {
    long n = i / 1728, k = i % 1728;
    WembT[i] = (_Float16)we[k*128 + n];
  }
  for (int i = tid; i < 384*128; i += stride) {
    int n = i >> 7, k = i & 127;
    WginT[i] = (_Float16)wgi[k*384 + n];
    WghhT[i] = (_Float16)wgh[k*384 + n];
  }
  for (int i = tid; i < 128*128; i += stride) {
    int n = i >> 7, k = i & 127;
    Wa1T[i] = (_Float16)wa1[k*128 + n];
    Wv1T[i] = (_Float16)wv1[k*128 + n];
  }
  for (int i = tid; i < 16*128; i += stride) {           // N pad 6->16 / 1->16
    int n = i >> 7, k = i & 127;
    Wa2T[i] = (_Float16)(n < 6 ? wa2[k*6 + n] : 0.f);
    Wv2T[i] = (_Float16)(n < 1 ? wv2[k] : 0.f);
  }
}

// ---------------------------------------------------------------- conv1-3 fused per-pixel MLP
// 4 waves * 16 rows/block over the [NPIX x 26] pixel matrix
__global__ void k_conv123(const float* __restrict__ obs,
                          const float* __restrict__ b1, const float* __restrict__ b2,
                          const float* __restrict__ b3,
                          const _Float16* __restrict__ W1T, const _Float16* __restrict__ W2T,
                          const _Float16* __restrict__ W3T,
                          _Float16* __restrict__ act3) {
  extern __shared__ char smem[];
  const int wave = threadIdx.x >> 5, lane = threadIdx.x & 31;
  _Float16* xa = (_Float16*)smem + wave * (16*32);            // 16x32 input staging
  _Float16* h1 = (_Float16*)smem + 4*16*32 + wave * (16*128); // 16x128 activation staging
  const long r0 = (long)blockIdx.x * 64 + wave * 16;
  const int col0 = lane & 15, radd = (lane >> 4) << 3;

  for (int i = lane; i < 16*32; i += 32) {
    int rr = i >> 5, cc = i & 31;
    xa[i] = (_Float16)((cc < 26) ? obs[(r0 + rr) * 26 + cc] : 0.f);
  }
  __syncthreads();

  v8f acc[8];
  { // layer1: [16x32]@[32x128]
    v16h A = load_a_rm(xa, 32);
#pragma unroll
    for (int nt = 0; nt < 8; ++nt)
      acc[nt] = wmma_f16(A, load_b_km(W1T + nt*16*32, 32), v8f_zero());
  }
#pragma unroll
  for (int nt = 0; nt < 8; ++nt) {
    int col = nt*16 + col0; float b = b1[col];
#pragma unroll
    for (int v = 0; v < 8; ++v) {
      float x = acc[nt][v] + b; x = x > 0.f ? x : 0.f;
      h1[(radd + v)*128 + col] = (_Float16)x;
    }
  }
  __syncthreads();
  v8f acc2[8];
#pragma unroll
  for (int nt = 0; nt < 8; ++nt) acc2[nt] = v8f_zero();
  for (int kt = 0; kt < 4; ++kt) { // layer2: [16x128]@[128x128]
    v16h A = load_a_rm(h1 + kt*32, 128);
#pragma unroll
    for (int nt = 0; nt < 8; ++nt)
      acc2[nt] = wmma_f16(A, load_b_km(W2T + nt*16*128 + kt*32, 128), acc2[nt]);
  }
  __syncthreads();
#pragma unroll
  for (int nt = 0; nt < 8; ++nt) {
    int col = nt*16 + col0; float b = b2[col];
#pragma unroll
    for (int v = 0; v < 8; ++v) {
      float x = acc2[nt][v] + b; x = x > 0.f ? x : 0.f;
      h1[(radd + v)*128 + col] = (_Float16)x;
    }
  }
  __syncthreads();
  v8f acc3 = v8f_zero();
  for (int kt = 0; kt < 4; ++kt)  // layer3: [16x128]@[128x16(pad)]
    acc3 = wmma_f16(load_a_rm(h1 + kt*32, 128), load_b_km(W3T + kt*32, 128), acc3);
  if (col0 < 8) {
    float b = b3[col0];
#pragma unroll
    for (int v = 0; v < 8; ++v) {
      float x = acc3[v] + b; x = x > 0.f ? x : 0.f;
      act3[(r0 + radd + v) * 8 + col0] = (_Float16)x;
    }
  }
}

// ---------------------------------------------------------------- conv4-6 (3x3 SAME), one image per block
__global__ void k_conv456(const _Float16* __restrict__ act3,
                          const float* __restrict__ b4, const float* __restrict__ b5,
                          const float* __restrict__ b6,
                          const _Float16* __restrict__ W4T, const _Float16* __restrict__ W5T,
                          const _Float16* __restrict__ W6T,
                          _Float16* __restrict__ cnnflat) {
  extern __shared__ char smem[];
  _Float16* pad3 = (_Float16*)smem;       // [8][11][8]  (halo=1)
  _Float16* pad4 = pad3 + 8*11*8;         // [8][11][16]
  _Float16* pad5 = pad4 + 8*11*16;        // [8][11][32]
  const int tid = threadIdx.x, wave = tid >> 5, lane = tid & 31;
  const long img = blockIdx.x;

  for (int i = tid; i < 8*11*(8+16+32); i += 128) pad3[i] = (_Float16)0.f;
  __syncthreads();
  for (int i = tid; i < PIX*8; i += 128) {
    int p = i >> 3, c = i & 7, y = p / 9, x = p % 9;
    pad3[((y+1)*11 + (x+1))*8 + c] = act3[img*PIX*8 + i];
  }
  __syncthreads();

  const int pix = wave*16 + (lane & 15);
  const bool valid = pix < PIX;
  const int py = valid ? pix / 9 : 0, px = valid ? pix % 9 : 0;
  const int col0 = lane & 15, radd = (lane >> 4) << 3;

  // conv4: im2col K=96, N=16
  v8f a4 = v8f_zero();
  for (int kt = 0; kt < 3; ++kt) {
    const int kb = kt * 32;
    v16h A = gather_a([&](int k) -> _Float16 {
      int kk = kb + k, patch = kk >> 3, ci = kk & 7;
      if (!valid || patch >= 9) return (_Float16)0.f;
      return pad3[((py + patch/3)*11 + (px + patch%3))*8 + ci];
    });
    a4 = wmma_f16(A, load_b_km(W4T + kb, 96), a4);
  }
  {
    float b = b4[col0];
#pragma unroll
    for (int v = 0; v < 8; ++v) {
      int p = wave*16 + radd + v;
      if (p < PIX) {
        int y = p/9, x = p%9;
        float xv = a4[v] + b; xv = xv > 0.f ? xv : 0.f;
        pad4[((y+1)*11 + (x+1))*16 + col0] = (_Float16)xv;
      }
    }
  }
  __syncthreads();

  // conv5: K=160, N=32
  v8f a5[2]; a5[0] = v8f_zero(); a5[1] = v8f_zero();
  for (int kt = 0; kt < 5; ++kt) {
    const int kb = kt * 32;
    v16h A = gather_a([&](int k) -> _Float16 {
      int kk = kb + k, patch = kk >> 4, ci = kk & 15;
      if (!valid || patch >= 9) return (_Float16)0.f;
      return pad4[((py + patch/3)*11 + (px + patch%3))*16 + ci];
    });
#pragma unroll
    for (int nt = 0; nt < 2; ++nt)
      a5[nt] = wmma_f16(A, load_b_km(W5T + nt*16*160 + kb, 160), a5[nt]);
  }
#pragma unroll
  for (int nt = 0; nt < 2; ++nt) {
    int col = nt*16 + col0; float b = b5[col];
#pragma unroll
    for (int v = 0; v < 8; ++v) {
      int p = wave*16 + radd + v;
      if (p < PIX) {
        int y = p/9, x = p%9;
        float xv = a5[nt][v] + b; xv = xv > 0.f ? xv : 0.f;
        pad5[((y+1)*11 + (x+1))*32 + col] = (_Float16)xv;
      }
    }
  }
  __syncthreads();

  // conv6: K=288 (one 32-ch patch per K step), N=32
  v8f a6[2]; a6[0] = v8f_zero(); a6[1] = v8f_zero();
  for (int kt = 0; kt < 9; ++kt) {
    const _Float16* src = pad5 + ((py + kt/3)*11 + (px + kt%3))*32;
    v16h A = gather_a([&](int k) -> _Float16 {
      return valid ? src[k] : (_Float16)0.f;
    });
#pragma unroll
    for (int nt = 0; nt < 2; ++nt)
      a6[nt] = wmma_f16(A, load_b_km(W6T + nt*16*288 + kt*32, 288), a6[nt]);
  }
#pragma unroll
  for (int nt = 0; nt < 2; ++nt) {
    int col = nt*16 + col0; float b = b6[col];
#pragma unroll
    for (int v = 0; v < 8; ++v) {
      int p = wave*16 + radd + v;
      if (p < PIX) {
        float xv = a6[nt][v] + b; xv = xv > 0.f ? xv : 0.f;
        cnnflat[img*1728 + p*32 + col] = (_Float16)xv;
      }
    }
  }
}

// ---------------------------------------------------------------- embedding dense + relu + LayerNorm
__global__ void k_emb_ln(const _Float16* __restrict__ cnnflat,
                         const _Float16* __restrict__ WembT,
                         const float* __restrict__ eb,
                         const float* __restrict__ lns, const float* __restrict__ lnb,
                         _Float16* __restrict__ emb) {
  extern __shared__ char smem[];
  const int wave = threadIdx.x >> 5, lane = threadIdx.x & 31;
  float* st = (float*)smem + wave * (16*128);
  float* mu = (float*)smem + 4*16*128 + wave*16;
  float* ri = (float*)smem + 4*16*128 + 64 + wave*16;
  const long r0 = (long)blockIdx.x * 64 + wave * 16;
  const int col0 = lane & 15, radd = (lane >> 4) << 3;

  v8f acc[8];
#pragma unroll
  for (int nt = 0; nt < 8; ++nt) acc[nt] = v8f_zero();
  const _Float16* arow = cnnflat + r0 * 1728;
  for (int kt = 0; kt < 54; ++kt) {
    v16h A = load_a_rm(arow + kt*32, 1728);
#pragma unroll
    for (int nt = 0; nt < 8; ++nt)
      acc[nt] = wmma_f16(A, load_b_km(WembT + (long)nt*16*1728 + kt*32, 1728), acc[nt]);
  }
#pragma unroll
  for (int nt = 0; nt < 8; ++nt) {
    int col = nt*16 + col0; float b = eb[col];
#pragma unroll
    for (int v = 0; v < 8; ++v) {
      float x = acc[nt][v] + b; x = x > 0.f ? x : 0.f;
      st[(radd+v)*128 + col] = x;
    }
  }
  __syncthreads();
  if (lane < 16) {
    float s = 0.f, s2 = 0.f;
    for (int c = 0; c < 128; ++c) { float x = st[lane*128 + c]; s += x; s2 += x*x; }
    float m = s * (1.f/128.f);
    float var = s2 * (1.f/128.f) - m*m;
    mu[lane] = m;
    ri[lane] = rsqrtf(var + 1e-6f);
  }
  __syncthreads();
  for (int i = lane; i < 16*128; i += 32) {
    int rr = i >> 7, c = i & 127;
    float x = (st[i] - mu[rr]) * ri[rr] * lns[c] + lnb[c];
    emb[(r0 + rr) * 128 + c] = (_Float16)x;
  }
}

// ---------------------------------------------------------------- GRU input gates for whole sequence
__global__ void k_igates(const _Float16* __restrict__ emb,
                         const _Float16* __restrict__ WginT,
                         const float* __restrict__ gb,
                         float* __restrict__ ig) {
  const int wave = threadIdx.x >> 5, lane = threadIdx.x & 31;
  const long r0 = (long)blockIdx.x * 64 + wave * 16;
  const int col0 = lane & 15, radd = (lane >> 4) << 3;
  const _Float16* arow = emb + r0 * 128;
  v16h Areg[4];
#pragma unroll
  for (int kt = 0; kt < 4; ++kt) Areg[kt] = load_a_rm(arow + kt*32, 128);
  for (int chunk = 0; chunk < 3; ++chunk) {
    v8f acc[8];
#pragma unroll
    for (int nt = 0; nt < 8; ++nt) acc[nt] = v8f_zero();
    for (int kt = 0; kt < 4; ++kt)
#pragma unroll
      for (int nt = 0; nt < 8; ++nt)
        acc[nt] = wmma_f16(Areg[kt], load_b_km(WginT + (chunk*128 + nt*16)*128 + kt*32, 128), acc[nt]);
#pragma unroll
    for (int nt = 0; nt < 8; ++nt) {
      int col = chunk*128 + nt*16 + col0;
      float b = gb[col];
#pragma unroll
      for (int v = 0; v < 8; ++v)
        ig[(r0 + radd + v) * 384 + col] = acc[nt][v] + b;
    }
  }
}

// ---------------------------------------------------------------- sequential GRU scan
// grid = 8 blocks x 1 wave; each block owns 16 batch rows of h for all T steps
__global__ void k_gru(const float* __restrict__ hidden0,
                      const unsigned char* __restrict__ dones,
                      const float* __restrict__ ig,
                      const _Float16* __restrict__ WghhT,
                      _Float16* __restrict__ outs,
                      float* __restrict__ final_h) {
  __shared__ float h[16*128];
  const int lane = threadIdx.x & 31;
  const int rb = blockIdx.x * 16;
  const int col0 = lane & 15, radd = (lane >> 4) << 3;
  for (int i = lane; i < 16*128; i += 32) h[i] = hidden0[(long)rb*128 + i];
  __syncthreads();

  for (int t = 0; t < T_DIM; ++t) {
    // reset rows whose episode ended
    for (int i = lane; i < 16*128; i += 32) {
      int rr = i >> 7;
      if (dones[t*B_DIM + rb + rr]) h[i] = 0.f;
    }
    __syncthreads();
    // h -> f16 A tiles
    v16h Areg[4];
    {
      const float* hr = h + (lane & 15) * 128;
#pragma unroll
      for (int kt = 0; kt < 4; ++kt) {
        const int kb = kt * 32;
        Areg[kt] = gather_a([&](int k) -> _Float16 { return (_Float16)hr[kb + k]; });
      }
    }
    // hg = h @ Whh, split z/r/n chunks of 128 cols
    v8f g[3][8];
#pragma unroll
    for (int c = 0; c < 3; ++c)
#pragma unroll
      for (int nt = 0; nt < 8; ++nt) g[c][nt] = v8f_zero();
#pragma unroll
    for (int c = 0; c < 3; ++c)
      for (int kt = 0; kt < 4; ++kt)
#pragma unroll
        for (int nt = 0; nt < 8; ++nt)
          g[c][nt] = wmma_f16(Areg[kt], load_b_km(WghhT + (c*128 + nt*16)*128 + kt*32, 128), g[c][nt]);
    // gating (each lane updates exactly its own (row,col) cells)
#pragma unroll
    for (int nt = 0; nt < 8; ++nt) {
      int col = nt*16 + col0;
#pragma unroll
      for (int v = 0; v < 8; ++v) {
        int rr = radd + v;
        long row = rb + rr;
        const float* igrow = ig + ((long)t * B_DIM + row) * 384;
        float z = 1.f / (1.f + __expf(-(igrow[col]       + g[0][nt][v])));
        float r = 1.f / (1.f + __expf(-(igrow[128 + col] + g[1][nt][v])));
        float n = tanhf(igrow[256 + col] + r * g[2][nt][v]);
        float hn = (1.f - z) * n + z * h[rr*128 + col];
        h[rr*128 + col] = hn;
        outs[((long)t * B_DIM + row) * 128 + col] = (_Float16)hn;
      }
    }
    __syncthreads();
  }
  for (int i = lane; i < 16*128; i += 32) final_h[(long)rb*128 + i] = h[i];
}

// ---------------------------------------------------------------- actor / critic heads
__global__ void k_heads(const _Float16* __restrict__ outs,
                        const _Float16* __restrict__ Wa1T, const float* __restrict__ a1b,
                        const _Float16* __restrict__ Wa2T, const float* __restrict__ a2b,
                        const _Float16* __restrict__ Wv1T, const float* __restrict__ v1b,
                        const _Float16* __restrict__ Wv2T, const float* __restrict__ v2b,
                        float* __restrict__ logits, float* __restrict__ value) {
  extern __shared__ char smem[];
  const int wave = threadIdx.x >> 5, lane = threadIdx.x & 31;
  _Float16* st = (_Float16*)smem + wave * (16*128);
  const long r0 = (long)blockIdx.x * 64 + wave * 16;
  const int col0 = lane & 15, radd = (lane >> 4) << 3;
  const _Float16* arow = outs + r0 * 128;
  v16h Areg[4];
#pragma unroll
  for (int kt = 0; kt < 4; ++kt) Areg[kt] = load_a_rm(arow + kt*32, 128);

  for (int head = 0; head < 2; ++head) {
    const _Float16* W1 = head ? Wv1T : Wa1T;
    const float*    B1 = head ? v1b  : a1b;
    const _Float16* W2 = head ? Wv2T : Wa2T;
    v8f acc[8];
#pragma unroll
    for (int nt = 0; nt < 8; ++nt) acc[nt] = v8f_zero();
    for (int kt = 0; kt < 4; ++kt)
#pragma unroll
      for (int nt = 0; nt < 8; ++nt)
        acc[nt] = wmma_f16(Areg[kt], load_b_km(W1 + nt*16*128 + kt*32, 128), acc[nt]);
    __syncthreads();
#pragma unroll
    for (int nt = 0; nt < 8; ++nt) {
      int col = nt*16 + col0; float b = B1[col];
#pragma unroll
      for (int v = 0; v < 8; ++v) {
        float x = acc[nt][v] + b; x = x > 0.f ? x : 0.f;
        st[(radd+v)*128 + col] = (_Float16)x;
      }
    }
    __syncthreads();
    v8f o = v8f_zero();
    for (int kt = 0; kt < 4; ++kt)
      o = wmma_f16(load_a_rm(st + kt*32, 128), load_b_km(W2 + kt*32, 128), o);
    if (head == 0) {
      if (col0 < 6) {
        float b = a2b[col0];
#pragma unroll
        for (int v = 0; v < 8; ++v)
          logits[(r0 + radd + v) * 6 + col0] = o[v] + b;
      }
    } else {
      if (col0 == 0) {
        float b = v2b[0];
#pragma unroll
        for (int v = 0; v < 8; ++v)
          value[r0 + radd + v] = o[v] + b;
      }
    }
    __syncthreads();
  }
}

// ----------------------------------------------------------------
extern "C" void kernel_launch(void* const* d_in, const int* in_sizes, int n_in,
                              void* d_out, int out_size, void* d_ws, size_t ws_size,
                              hipStream_t stream) {
  (void)in_sizes; (void)n_in; (void)out_size; (void)ws_size;
  const float* hidden = (const float*)d_in[0];
  const float* obs    = (const float*)d_in[1];
  const unsigned char* dones = (const unsigned char*)d_in[2];
  const float* c1w = (const float*)d_in[3];  const float* c1b = (const float*)d_in[4];
  const float* c2w = (const float*)d_in[5];  const float* c2b = (const float*)d_in[6];
  const float* c3w = (const float*)d_in[7];  const float* c3b = (const float*)d_in[8];
  const float* c4w = (const float*)d_in[9];  const float* c4b = (const float*)d_in[10];
  const float* c5w = (const float*)d_in[11]; const float* c5b = (const float*)d_in[12];
  const float* c6w = (const float*)d_in[13]; const float* c6b = (const float*)d_in[14];
  const float* embw = (const float*)d_in[15]; const float* embb = (const float*)d_in[16];
  const float* lns = (const float*)d_in[17]; const float* lnb = (const float*)d_in[18];
  const float* giw = (const float*)d_in[19]; const float* gib = (const float*)d_in[20];
  const float* ghw = (const float*)d_in[21];
  const float* a1w = (const float*)d_in[22]; const float* a1b = (const float*)d_in[23];
  const float* a2w = (const float*)d_in[24]; const float* a2b = (const float*)d_in[25];
  const float* v1w = (const float*)d_in[26]; const float* v1b = (const float*)d_in[27];
  const float* v2w = (const float*)d_in[28]; const float* v2b = (const float*)d_in[29];

  char* ws = (char*)d_ws;
  auto carve = [&](size_t bytes) -> void* {
    void* p = (void*)ws;
    ws += (bytes + 255) & ~(size_t)255;
    return p;
  };
  _Float16* W1T   = (_Float16*)carve(128*32*2);
  _Float16* W2T   = (_Float16*)carve(128*128*2);
  _Float16* W3T   = (_Float16*)carve(16*128*2);
  _Float16* W4T   = (_Float16*)carve(16*96*2);
  _Float16* W5T   = (_Float16*)carve(32*160*2);
  _Float16* W6T   = (_Float16*)carve(32*288*2);
  _Float16* WembT = (_Float16*)carve(128UL*1728*2);
  _Float16* WginT = (_Float16*)carve(384UL*128*2);
  _Float16* WghhT = (_Float16*)carve(384UL*128*2);
  _Float16* Wa1T  = (_Float16*)carve(128*128*2);
  _Float16* Wa2T  = (_Float16*)carve(16*128*2);
  _Float16* Wv1T  = (_Float16*)carve(128*128*2);
  _Float16* Wv2T  = (_Float16*)carve(16*128*2);
  _Float16* act3    = (_Float16*)carve((size_t)TB*PIX*8*2);
  _Float16* cnnflat = (_Float16*)carve((size_t)TB*1728*2);
  _Float16* embA    = (_Float16*)carve((size_t)TB*128*2);
  float*    ig      = (float*)   carve((size_t)TB*384*4);
  _Float16* outsA   = (_Float16*)carve((size_t)TB*128*2);

  float* out = (float*)d_out;
  float* final_h = out;                 // [128,128]
  float* logits  = out + TB;            // [T,B,6]
  float* value   = out + TB + TB*6;     // [T,B]

  k_prep<<<256, 256, 0, stream>>>(c1w, c2w, c3w, c4w, c5w, c6w, embw, giw, ghw,
                                  a1w, a2w, v1w, v2w,
                                  W1T, W2T, W3T, W4T, W5T, W6T,
                                  WembT, WginT, WghhT, Wa1T, Wa2T, Wv1T, Wv2T);

  size_t sm1 = (4*16*32 + 4*16*128) * sizeof(_Float16);
  k_conv123<<<NPIX/64, 128, sm1, stream>>>(obs, c1b, c2b, c3b, W1T, W2T, W3T, act3);

  size_t sm2 = (size_t)(8*11*(8+16+32)) * sizeof(_Float16);
  k_conv456<<<TB, 128, sm2, stream>>>(act3, c4b, c5b, c6b, W4T, W5T, W6T, cnnflat);

  size_t sm3 = (4*16*128) * sizeof(float) + 128 * sizeof(float);
  k_emb_ln<<<TB/64, 128, sm3, stream>>>(cnnflat, WembT, embb, lns, lnb, embA);

  k_igates<<<TB/64, 128, 0, stream>>>(embA, WginT, gib, ig);

  k_gru<<<8, 32, 0, stream>>>(hidden, dones, ig, WghhT, outsA, final_h);

  size_t sm4 = (4*16*128) * sizeof(_Float16);
  k_heads<<<TB/64, 128, sm4, stream>>>(outsA, Wa1T, a1b, Wa2T, a2b,
                                       Wv1T, v1b, Wv2T, v2b, logits, value);
}
